// SelfAttention_33294586478799
// MI455X (gfx1250) — compile-verified, matrix operand order checked by reading, not run on hardware
//
#include <hip/hip_runtime.h>
#include <hip/hip_bf16.h>

typedef __attribute__((ext_vector_type(16))) __bf16 v16bf;
typedef __attribute__((ext_vector_type(8)))  __bf16 v8bf;
typedef __attribute__((ext_vector_type(8)))  float  v8f;

#define HEADS 16
#define DH    64
#define DM    1024
#define BATCH 4
#define SEQ   2048

// ---------------------------------------------------------------------------
// WMMA helper: D = A(16x32 bf16) * B(32x16 bf16) + C(16x16 f32)
// ---------------------------------------------------------------------------
__device__ __forceinline__ v8f wmma_bf16(v16bf a, v16bf b, v8f c) {
  return __builtin_amdgcn_wmma_f32_16x16x32_bf16(
      /*neg_a=*/false, a, /*neg_b=*/false, b,
      /*c_mod=*/(short)0, c, /*reuse_a=*/false, /*reuse_b=*/false);
}

// ---------------------------------------------------------------------------
// Fragment loaders: documented gfx1250 16-bit A/B layout.
// Lane L holds row/col (L & 15); K runs: base=(L>>4)*8, elements [base..base+7]
// and [base+16..base+23].  base pointer must be 16B aligned; ld in elements.
// ---------------------------------------------------------------------------
__device__ __forceinline__ v16bf load_frag_bf16(const __bf16* base, int ld) {
  const int lane = threadIdx.x & 31;
  const __bf16* p = base + (size_t)(lane & 15) * ld + ((lane >> 4) << 3);
  v8bf lo = *(const v8bf*)(p);
  v8bf hi = *(const v8bf*)(p + 16);
  v16bf r;
#pragma unroll
  for (int i = 0; i < 8; ++i) { r[i] = lo[i]; r[i + 8] = hi[i]; }
  return r;
}

__device__ __forceinline__ v16bf load_frag_f32(const float* base, int ld) {
  const int lane = threadIdx.x & 31;
  const float* p = base + (size_t)(lane & 15) * ld + ((lane >> 4) << 3);
  v16bf r;
#pragma unroll
  for (int i = 0; i < 8; ++i) {
    r[i]     = (__bf16)p[i];
    r[i + 8] = (__bf16)p[i + 16];
  }
  return r;
}

// 16-lane-group reductions (wave32: two independent groups of 16 lanes)
__device__ __forceinline__ float redmax16(float v) {
  v = fmaxf(v, __shfl_xor(v, 1, 16));
  v = fmaxf(v, __shfl_xor(v, 2, 16));
  v = fmaxf(v, __shfl_xor(v, 4, 16));
  v = fmaxf(v, __shfl_xor(v, 8, 16));
  return v;
}
__device__ __forceinline__ float redsum16(float v) {
  v += __shfl_xor(v, 1, 16);
  v += __shfl_xor(v, 2, 16);
  v += __shfl_xor(v, 4, 16);
  v += __shfl_xor(v, 8, 16);
  return v;
}

// ---------------------------------------------------------------------------
// Kernel 0: convert weights fp32 -> bf16 (Wq|Wk|Wv packed, Wp)
// ---------------------------------------------------------------------------
__global__ void __launch_bounds__(256) cvt_weights_kernel(
    const float* __restrict__ Wq, const float* __restrict__ Wk,
    const float* __restrict__ Wv, const float* __restrict__ Wp,
    __bf16* __restrict__ wqkv, __bf16* __restrict__ wpb) {
  const int stride = gridDim.x * blockDim.x;
  const int i0 = blockIdx.x * blockDim.x + threadIdx.x;
  for (int i = i0; i < 3 * DH * DH; i += stride) {
    const float* s = (i < DH * DH) ? Wq : (i < 2 * DH * DH) ? Wk : Wv;
    wqkv[i] = (__bf16)s[i % (DH * DH)];
  }
  for (int i = i0; i < DM * DM; i += stride) wpb[i] = (__bf16)Wp[i];
}

// ---------------------------------------------------------------------------
// Kernel 1: per-head QKV projection.  One wave -> one 16(t) x 64(e) tile.
// Q,K stored [B,H,T,64]; V stored transposed [B,H,64,T] for the P@V stage.
// ---------------------------------------------------------------------------
__global__ void __launch_bounds__(32) qkv_proj_kernel(
    const float* __restrict__ x, const __bf16* __restrict__ wqkv,
    const float* __restrict__ bq, const float* __restrict__ bk,
    const float* __restrict__ bv,
    __bf16* __restrict__ qk_out, __bf16* __restrict__ vt_out) {
  const int sel  = blockIdx.y;                 // 0=q 1=k 2=v
  const int nt16 = SEQ / 16;
  const int t0   = (blockIdx.x % nt16) * 16;
  const int bh   = blockIdx.x / nt16;
  const int lane = threadIdx.x & 31;
  const int n    = lane & 15;
  const int hig  = lane >> 4;
  const int b    = bh / HEADS, h = bh % HEADS;

  const float* xp = x + ((size_t)(b * SEQ + t0)) * DM + h * DH;
  v16bf a0 = load_frag_f32(xp, DM);        // k = 0..31 of head dim
  v16bf a1 = load_frag_f32(xp + 32, DM);   // k = 32..63

  const __bf16* W    = wqkv + sel * (DH * DH);
  const float*  bias = (sel == 0) ? bq : (sel == 1) ? bk : bv;

  v8f acc[4];
#pragma unroll
  for (int c = 0; c < 4; ++c) {
    const float bval = bias[c * 16 + n];
#pragma unroll
    for (int j = 0; j < 8; ++j) acc[c][j] = bval;
    v16bf w0 = load_frag_bf16(W + (size_t)(c * 16) * DH, DH);
    v16bf w1 = load_frag_bf16(W + (size_t)(c * 16) * DH + 32, DH);
    acc[c] = wmma_bf16(a0, w0, acc[c]);
    acc[c] = wmma_bf16(a1, w1, acc[c]);
  }

  if (sel < 2) {
    __bf16* out = qk_out + (size_t)sel * ((size_t)BATCH * HEADS * SEQ * DH) +
                  ((size_t)(bh * SEQ + t0)) * DH;
#pragma unroll
    for (int c = 0; c < 4; ++c)
#pragma unroll
      for (int j = 0; j < 8; ++j)
        out[(size_t)(j + 8 * hig) * DH + c * 16 + n] = (__bf16)acc[c][j];
  } else {
    __bf16* vt = vt_out + (size_t)bh * DH * SEQ;
#pragma unroll
    for (int c = 0; c < 4; ++c)
#pragma unroll
      for (int j = 0; j < 8; ++j)
        vt[(size_t)(c * 16 + n) * SEQ + t0 + j + 8 * hig] = (__bf16)acc[c][j];
  }
}

// ---------------------------------------------------------------------------
// Kernel 2: flash attention with double-buffered LDS staging of K/V.
// 8 waves/block, wave owns 16 query rows; the block cooperatively stages each
// 32-key K chunk (4KB) and V^T chunk (4KB) into LDS once (8x traffic cut vs
// per-wave global reads).  Global loads for chunk kb+1 are issued into regs
// before the WMMAs for chunk kb (latency hidden), stored to the alternate
// buffer after compute; one barrier per iteration.
// ---------------------------------------------------------------------------
__global__ void __launch_bounds__(256) flash_attn_kernel(
    const __bf16* __restrict__ Q, const __bf16* __restrict__ K,
    const __bf16* __restrict__ Vt, __bf16* __restrict__ O) {
  __shared__ __bf16 Kl[2][32 * DH];     // 2 x 4KB : 32 keys x 64 d
  __shared__ __bf16 Vl[2][DH * 32];     // 2 x 4KB : 64 d x 32 keys (V^T)
  __shared__ __bf16 plds[8][16 * 32];   // 8KB, one 16x32 P slice per wave

  const int tid  = threadIdx.x;
  const int wave = tid >> 5;
  const int lane = tid & 31;
  const int n = lane & 15, hig = lane >> 4;
  const int bh = blockIdx.y;
  const int b = bh / HEADS, h = bh % HEADS;
  const int q0 = blockIdx.x * 128 + wave * 16;

  const __bf16* Qp = Q + ((size_t)(bh * SEQ + q0)) * DH;
  const v16bf qa0 = load_frag_bf16(Qp, DH);
  const v16bf qa1 = load_frag_bf16(Qp + 32, DH);

  const __bf16* Kb = K + (size_t)bh * SEQ * DH;
  const __bf16* Vb = Vt + (size_t)bh * DH * SEQ;
  __bf16*      myl = plds[wave];

  // cooperative-stage index split: 256 threads x 8 bf16 = 2048 elems (4KB)
  const int vrow = tid >> 2, vpart = tid & 3;

  v8f   o[4];
  float rmax[8], rsum[8];
#pragma unroll
  for (int c = 0; c < 4; ++c)
#pragma unroll
    for (int j = 0; j < 8; ++j) o[c][j] = 0.0f;
#pragma unroll
  for (int j = 0; j < 8; ++j) { rmax[j] = -3.0e38f; rsum[j] = 0.0f; }

  const float SC = 0.03125f * 1.44269504088896f;  // (1/sqrt(C)) * log2(e)
  const int NCH = SEQ / 32;

  // prologue: stage chunk 0 into buffer 0
  {
    v8bf kr = *(const v8bf*)(Kb + tid * 8);
    v8bf vr = *(const v8bf*)(Vb + (size_t)vrow * SEQ + vpart * 8);
    *(v8bf*)(&Kl[0][tid * 8]) = kr;
    *(v8bf*)(&Vl[0][vrow * 32 + vpart * 8]) = vr;
  }
  __syncthreads();

  for (int kb = 0; kb < NCH; ++kb) {
    const int cur = kb & 1;
    const bool more = (kb + 1 < NCH);

    // issue global loads for chunk kb+1 (in flight during the WMMAs below)
    v8bf kr{}, vr{};
    if (more) {
      kr = *(const v8bf*)(Kb + (size_t)((kb + 1) * 32) * DH + tid * 8);
      vr = *(const v8bf*)(Vb + (size_t)vrow * SEQ + (kb + 1) * 32 + vpart * 8);
      if (kb + 2 < NCH)  // global_prefetch for chunk kb+2
        __builtin_prefetch(Kb + (size_t)((kb + 2) * 32) * DH, 0, 0);
    }

    // S[16q x 32k] = Q @ K^T from LDS : two 16x16 C-tiles, 2 WMMAs each
    const __bf16* Kc = Kl[cur];
    v8f s0, s1;
#pragma unroll
    for (int j = 0; j < 8; ++j) { s0[j] = 0.0f; s1[j] = 0.0f; }
    {
      v16bf k00 = load_frag_bf16(Kc, DH);
      v16bf k01 = load_frag_bf16(Kc + 32, DH);
      v16bf k10 = load_frag_bf16(Kc + 16 * DH, DH);
      v16bf k11 = load_frag_bf16(Kc + 16 * DH + 32, DH);
      s0 = wmma_bf16(qa0, k00, s0);
      s0 = wmma_bf16(qa1, k01, s0);
      s1 = wmma_bf16(qa0, k10, s1);
      s1 = wmma_bf16(qa1, k11, s1);
    }

    // online softmax (log2 domain); row r's 16 N-elems live in one 16-lane
    // group at vgpr j, r = j + 8*(lane>=16)
#pragma unroll
    for (int j = 0; j < 8; ++j) {
      const float a  = s0[j] * SC;
      const float bb = s1[j] * SC;
      const float cm = redmax16(fmaxf(a, bb));
      const float mn = fmaxf(rmax[j], cm);
      const float p0 = __builtin_amdgcn_exp2f(a - mn);
      const float p1 = __builtin_amdgcn_exp2f(bb - mn);
      const float f  = __builtin_amdgcn_exp2f(rmax[j] - mn);
      rsum[j] = rsum[j] * f + redsum16(p0 + p1);
      rmax[j] = mn;
#pragma unroll
      for (int c = 0; c < 4; ++c) o[c][j] *= f;
      const int row = j + 8 * hig;
      myl[row * 32 + n]      = (__bf16)p0;   // C-layout -> row-major LDS
      myl[row * 32 + 16 + n] = (__bf16)p1;
    }

    // P back as A-operand (DS ops are wave-ordered; private slice, no barrier)
    const v16bf pa = load_frag_bf16(myl, 32);

    // O += P @ V from LDS (V^T: columns contiguous in key dim)
    const __bf16* Vc = Vl[cur];
#pragma unroll
    for (int c = 0; c < 4; ++c) {
      v16bf vf = load_frag_bf16(Vc + (c * 16) * 32, 32);
      o[c] = wmma_bf16(pa, vf, o[c]);
    }

    // store staged chunk kb+1 to the alternate buffer (its readers finished
    // before the previous barrier), then make it visible
    if (more) {
      *(v8bf*)(&Kl[1 - cur][tid * 8]) = kr;
      *(v8bf*)(&Vl[1 - cur][vrow * 32 + vpart * 8]) = vr;
    }
    __syncthreads();
  }

  // normalize and store bf16 to [B,T,C] (head-interleaved for the projection)
  __bf16* op = O + ((size_t)(b * SEQ + q0)) * DM + h * DH;
#pragma unroll
  for (int j = 0; j < 8; ++j) {
    const float inv = 1.0f / rsum[j];
    const int row = j + 8 * hig;
#pragma unroll
    for (int c = 0; c < 4; ++c)
      op[(size_t)row * DM + c * 16 + n] = (__bf16)(o[c][j] * inv);
  }
}

// ---------------------------------------------------------------------------
// Kernel 3: output projection  Y = O @ Wp^T + bp  (M=8192, N=K=1024, fp32 out)
// 8 waves/block -> 128x64 block tile.  The Wp k-slice (64 cols x 32 k, 4KB)
// is shared by all 8 waves -> double-buffered LDS staging (8x traffic cut).
// ---------------------------------------------------------------------------
__global__ void __launch_bounds__(256) out_proj_kernel(
    const __bf16* __restrict__ Ob, const __bf16* __restrict__ Wpb,
    const float* __restrict__ bp, float* __restrict__ out) {
  __shared__ __bf16 Wl[2][64 * 32];     // 2 x 4KB

  const int tid  = threadIdx.x;
  const int wave = tid >> 5;
  const int lane = tid & 31;
  const int n = lane & 15, hig = lane >> 4;
  const int r0 = blockIdx.x * 128 + wave * 16;
  const int n0 = blockIdx.y * 64;
  const int wrow = tid >> 2, wpart = tid & 3;

  v8f acc[4];
#pragma unroll
  for (int c = 0; c < 4; ++c) {
    const float bval = bp[n0 + c * 16 + n];
#pragma unroll
    for (int j = 0; j < 8; ++j) acc[c][j] = bval;
  }

  // prologue: stage Wp k-slice 0
  {
    v8bf wr = *(const v8bf*)(Wpb + (size_t)(n0 + wrow) * DM + wpart * 8);
    *(v8bf*)(&Wl[0][wrow * 32 + wpart * 8]) = wr;
  }
  __syncthreads();

  const int NKC = DM / 32;
  for (int kc = 0; kc < NKC; ++kc) {
    const int cur = kc & 1;
    const bool more = (kc + 1 < NKC);

    v8bf wr{};
    if (more)
      wr = *(const v8bf*)(Wpb + (size_t)(n0 + wrow) * DM + (kc + 1) * 32 + wpart * 8);

    v16bf a = load_frag_bf16(Ob + (size_t)r0 * DM + kc * 32, DM);
#pragma unroll
    for (int c = 0; c < 4; ++c) {
      v16bf w = load_frag_bf16(&Wl[cur][(c * 16) * 32], 32);
      acc[c] = wmma_bf16(a, w, acc[c]);
    }

    if (more)
      *(v8bf*)(&Wl[1 - cur][wrow * 32 + wpart * 8]) = wr;
    __syncthreads();
  }

#pragma unroll
  for (int c = 0; c < 4; ++c)
#pragma unroll
    for (int j = 0; j < 8; ++j)
      out[(size_t)(r0 + j + 8 * hig) * DM + n0 + c * 16 + n] = acc[c][j];
}

// ---------------------------------------------------------------------------
// Host launcher
// ---------------------------------------------------------------------------
extern "C" void kernel_launch(void* const* d_in, const int* in_sizes, int n_in,
                              void* d_out, int out_size, void* d_ws, size_t ws_size,
                              hipStream_t stream) {
  const float* x  = (const float*)d_in[0];
  const float* Wq = (const float*)d_in[1];
  const float* bq = (const float*)d_in[2];
  const float* Wk = (const float*)d_in[3];
  const float* bk = (const float*)d_in[4];
  const float* Wv = (const float*)d_in[5];
  const float* bv = (const float*)d_in[6];
  const float* Wp = (const float*)d_in[7];
  const float* bp = (const float*)d_in[8];
  float* out = (float*)d_out;

  __bf16* ws = (__bf16*)d_ws;
  const size_t QKV_E = (size_t)BATCH * HEADS * SEQ * DH;  // 8,388,608 elems each
  __bf16* Qw   = ws;                       // [B,H,T,64]   bf16
  __bf16* Vtw  = ws + 2 * QKV_E;           // [B,H,64,T]   bf16 (transposed V)
  __bf16* Ow   = ws + 3 * QKV_E;           // [B,T,1024]   bf16
  __bf16* Wpb  = ws + 4 * QKV_E;           // [1024,1024]  bf16
  __bf16* Wqkv = Wpb + (size_t)DM * DM;    // 3 x [64,64]  bf16
  // total: ~69.2 MB of d_ws

  cvt_weights_kernel<<<256, 256, 0, stream>>>(Wq, Wk, Wv, Wp, Wqkv, Wpb);

  dim3 g1(BATCH * HEADS * (SEQ / 16), 3);
  qkv_proj_kernel<<<g1, 32, 0, stream>>>(x, Wqkv, bq, bk, bv, Qw, Vtw);

  dim3 g2(SEQ / 128, BATCH * HEADS);
  flash_attn_kernel<<<g2, 256, 0, stream>>>(Qw, Qw + QKV_E, Vtw, Ow);

  dim3 g3((BATCH * SEQ) / 128, DM / 64);
  out_proj_kernel<<<g3, 256, 0, stream>>>(Ow, Wpb, bp, out);
}
